// Hasher_60370060312718
// MI455X (gfx1250) — matrix operand order repaired and checked by reading, not compiled, not run on hardware
//
#include <hip/hip_runtime.h>
#include <hip/hip_bf16.h>

typedef __attribute__((ext_vector_type(16))) _Float16 v16h;
typedef __attribute__((ext_vector_type(8)))  float    v8f;

#define NEL 589824   // 256*256*3*3
#define DD  64
#define NBIN 64
#define HID 128

// workspace layout (float indices)
#define WS_HVEC  0                  // 64 floats
#define WS_BINS  64                 // 64 floats
#define WS_CVEC  128                // 128 floats
#define WS_BPACK 256                // 8192 halfs (4096 float slots)
#define WS_PART  (256 + 4096)       // 256*64 floats of partial sums

// output layout (floats): new_weight[N], x_up[N], Q[N*64], prob[N*64], bins[64]
#define OFF_NW   0L
#define OFF_XUP  ((long)NEL)
#define OFF_Q    (2L * NEL)
#define OFF_PROB (66L * NEL)
#define OFF_BINS (130L * NEL)

// ---------------- Kernel 1: partial sums of x . w_down (deterministic 2-pass)
__global__ __launch_bounds__(256) void k_down_partial(
    const float* __restrict__ x, const float* __restrict__ w_down,
    float* __restrict__ part) {
  __shared__ float red[16][64];
  int tid = threadIdx.x;
  int c4 = (tid & 15) * 4;      // 4 consecutive cols
  int g  = tid >> 4;            // 16 rows in flight
  long row0 = (long)blockIdx.x * 2304;
  float4 acc = make_float4(0.f, 0.f, 0.f, 0.f);
  for (int i = 0; i < 144; ++i) {
    long r = row0 + (long)i * 16 + g;
    float xv = x[r];
    float4 wv = *(const float4*)(w_down + r * 64 + c4);
    acc.x += xv * wv.x; acc.y += xv * wv.y;
    acc.z += xv * wv.z; acc.w += xv * wv.w;
  }
  red[g][c4 + 0] = acc.x; red[g][c4 + 1] = acc.y;
  red[g][c4 + 2] = acc.z; red[g][c4 + 3] = acc.w;
  __syncthreads();
  if (tid < 64) {
    float s = 0.f;
    #pragma unroll
    for (int g2 = 0; g2 < 16; ++g2) s += red[g2][tid];
    part[blockIdx.x * 64 + tid] = s;
  }
}

// ---------------- Kernel 2: h, bins, cvec, and WMMA-layout f16 pack of w_c2
__global__ __launch_bounds__(256) void k_finalize(
    const float* __restrict__ part, const float* __restrict__ b_down,
    const float* __restrict__ w_hash, const float* __restrict__ b_hash,
    const float* __restrict__ w_c1, const float* __restrict__ b_c1,
    const float* __restrict__ w_c2, float* __restrict__ ws,
    float* __restrict__ out_bins) {
  __shared__ float sh[64];
  __shared__ float sb[64];
  int tid = threadIdx.x;
  if (tid < 64) {
    float s = b_down[tid];
    for (int b = 0; b < 256; ++b) s += part[b * 64 + tid];
    s = fmaxf(s, 0.f);
    ws[WS_HVEC + tid] = s;
    sh[tid] = s;
  }
  __syncthreads();
  if (tid < 64) {
    float s = b_hash[tid];
    #pragma unroll 8
    for (int d = 0; d < 64; ++d) s += sh[d] * w_hash[d * 64 + tid];
    ws[WS_BINS + tid] = s;
    sb[tid] = s;
    out_bins[tid] = s;
  }
  __syncthreads();
  if (tid < 128) {
    float s = b_c1[tid];
    #pragma unroll 8
    for (int j = 0; j < 64; ++j) s += sb[j] * w_c1[j * 128 + tid];
    ws[WS_CVEC + tid] = s;
  }
  // pack w_c2 [128,64] into B fragments for v_wmma_f32_16x16x32_f16:
  // fragment f = kt*4+nt, lane L, elem e:  K = kt*32 + (L<16?0:16) + e,
  //                                        N = nt*16 + (L&15)
  _Float16* bp = (_Float16*)(ws + WS_BPACK);
  for (int t = 0; t < 32; ++t) {
    int idx = tid * 32 + t;        // 0..8191
    int f = idx >> 9;
    int L = (idx >> 4) & 31;
    int e = idx & 15;
    int kt = f >> 2, nt = f & 3;
    int k = kt * 32 + ((L < 16) ? 0 : 16) + e;
    int n = nt * 16 + (L & 15);
    bp[idx] = (_Float16)w_c2[k * 64 + n];
  }
}

// ---------------- Kernel 3: x_up = h . w_up + b_up (streaming float4)
__global__ __launch_bounds__(256) void k_up(
    const float* __restrict__ w_up, const float* __restrict__ b_up,
    const float* __restrict__ ws, float* __restrict__ out_xup) {
  __shared__ float sh[64];
  int tid = threadIdx.x;
  if (tid < 64) sh[tid] = ws[WS_HVEC + tid];
  __syncthreads();
  long n = ((long)blockIdx.x * 256 + tid) * 4;
  float4 acc = *(const float4*)(b_up + n);
  #pragma unroll 4
  for (int d = 0; d < 64; ++d) {
    float4 wv = *(const float4*)(w_up + (long)d * NEL + n);
    float hd = sh[d];
    acc.x += hd * wv.x; acc.y += hd * wv.y;
    acc.z += hd * wv.z; acc.w += hd * wv.w;
  }
  *(float4*)(out_xup + n) = acc;
}

// ---------------- Kernel 4: fused CentroidNet (WMMA) + softmax + prob/Q/new_weight
__global__ __launch_bounds__(256) void k_centroid(
    const float* __restrict__ x, const float* __restrict__ ws,
    const float* __restrict__ w_c1, const float* __restrict__ b_c2,
    float* __restrict__ out) {
  __shared__ v16h  ldsBv[512];          // 16 fragments x 32 lanes (16 KB)
  __shared__ float scv[128];
  __shared__ float suv[128];
  __shared__ float sbins[64];
  __shared__ float sbc2[64];
  __shared__ float slog[8][16][68];     // per-wave logits, padded stride

  int tid = threadIdx.x;
  if (tid < 128) scv[tid] = ws[WS_CVEC + tid];
  if (tid < 128) suv[tid] = w_c1[64 * 128 + tid];   // last row of w_c1
  if (tid < 64)  sbins[tid] = ws[WS_BINS + tid];
  if (tid < 64)  sbc2[tid] = b_c2[tid];
  {
    const float4* src = (const float4*)(ws + WS_BPACK);
    float4* dst = (float4*)ldsBv;
    #pragma unroll
    for (int i = 0; i < 4; ++i) dst[tid + 256 * i] = src[tid + 256 * i];
  }
  __syncthreads();

  int wv = tid >> 5, lane = tid & 31;
  long mbase = (long)blockIdx.x * 128 + wv * 16;
  float xm = x[mbase + (lane & 15)];
  int abase = (lane < 16) ? 0 : 16;

  v8f acc0 = {}, acc1 = {}, acc2 = {}, acc3 = {};
  const _Float16* lb = (const _Float16*)ldsBv;
  #pragma unroll
  for (int kt = 0; kt < 4; ++kt) {
    v16h a;
    #pragma unroll
    for (int e = 0; e < 16; ++e) {
      // A k order used here (same as B pack): lane<16 -> K = e; lane>=16 -> K = 16 + e
      int k = kt * 32 + abase + e;
      float hval = scv[k] + xm * suv[k];
      a[e] = (_Float16)fmaxf(hval, 0.f);
    }
    v16h b0 = *(const v16h*)(lb + ((kt * 4 + 0) * 32 + lane) * 16);
    v16h b1 = *(const v16h*)(lb + ((kt * 4 + 1) * 32 + lane) * 16);
    v16h b2 = *(const v16h*)(lb + ((kt * 4 + 2) * 32 + lane) * 16);
    v16h b3 = *(const v16h*)(lb + ((kt * 4 + 3) * 32 + lane) * 16);
    acc0 = __builtin_amdgcn_wmma_f32_16x16x32_f16(false, a, false, b0, (short)0, acc0, false, false);
    acc1 = __builtin_amdgcn_wmma_f32_16x16x32_f16(false, a, false, b1, (short)0, acc1, false, false);
    acc2 = __builtin_amdgcn_wmma_f32_16x16x32_f16(false, a, false, b2, (short)0, acc2, false, false);
    acc3 = __builtin_amdgcn_wmma_f32_16x16x32_f16(false, a, false, b3, (short)0, acc3, false, false);
  }

  // scatter logits (+bias) to LDS: C layout VGPR r -> M = r (+8 for hi lanes), N = nt*16 + lane%16
  int col0 = lane & 15;
  int ml = (lane < 16) ? 0 : 8;
  #pragma unroll
  for (int r = 0; r < 8; ++r) {
    slog[wv][r + ml][ 0 + col0] = acc0[r] + sbc2[ 0 + col0];
    slog[wv][r + ml][16 + col0] = acc1[r] + sbc2[16 + col0];
    slog[wv][r + ml][32 + col0] = acc2[r] + sbc2[32 + col0];
    slog[wv][r + ml][48 + col0] = acc3[r] + sbc2[48 + col0];
  }
  __syncthreads();

  // softmax: 2 lanes per row (row m = lane%16, half hh = lane>>4)
  int m = lane & 15;
  int hh = lane >> 4;
  const float* row = &slog[wv][m][0];
  float mx = -3.0e38f;
  #pragma unroll
  for (int j = 0; j < 32; ++j) mx = fmaxf(mx, row[hh * 32 + j]);
  mx = fmaxf(mx, __shfl_xor(mx, 16, 32));
  float ev[32];
  float sum = 0.f;
  #pragma unroll
  for (int j = 0; j < 32; ++j) {
    float e = __expf(row[hh * 32 + j] - mx);
    ev[j] = e; sum += e;
  }
  sum += __shfl_xor(sum, 16, 32);
  float inv = 1.0f / sum;

  long mg = mbase + m;
  float nw = 0.f;
  float4* pout = (float4*)(out + OFF_PROB + mg * 64 + hh * 32);
  #pragma unroll
  for (int j = 0; j < 32; j += 4) {
    int n = hh * 32 + j;
    float p0 = ev[j] * inv, p1 = ev[j + 1] * inv;
    float p2 = ev[j + 2] * inv, p3 = ev[j + 3] * inv;
    nw += sbins[n] * p0 + sbins[n + 1] * p1 + sbins[n + 2] * p2 + sbins[n + 3] * p3;
    pout[j >> 2] = make_float4(p0, p1, p2, p3);
  }
  nw += __shfl_xor(nw, 16, 32);
  if (hh == 0) out[OFF_NW + mg] = nw * (1.0f / 64.0f);

  // Q = normalize(1/(x - bins + 1))
  float qv[32];
  float qs = 0.f;
  #pragma unroll
  for (int j = 0; j < 32; ++j) {
    float q = 1.0f / (xm - sbins[hh * 32 + j] + 1.0f);
    qv[j] = q; qs += q;
  }
  qs += __shfl_xor(qs, 16, 32);
  float qinv = 1.0f / qs;
  float4* qout = (float4*)(out + OFF_Q + mg * 64 + hh * 32);
  #pragma unroll
  for (int j = 0; j < 32; j += 4)
    qout[j >> 2] = make_float4(qv[j] * qinv, qv[j + 1] * qinv,
                               qv[j + 2] * qinv, qv[j + 3] * qinv);
}

extern "C" void kernel_launch(void* const* d_in, const int* in_sizes, int n_in,
                              void* d_out, int out_size, void* d_ws, size_t ws_size,
                              hipStream_t stream) {
  const float* x      = (const float*)d_in[0];
  const float* w_down = (const float*)d_in[1];
  const float* b_down = (const float*)d_in[2];
  const float* w_up   = (const float*)d_in[3];
  const float* b_up   = (const float*)d_in[4];
  const float* w_hash = (const float*)d_in[5];
  const float* b_hash = (const float*)d_in[6];
  const float* w_c1   = (const float*)d_in[7];
  const float* b_c1   = (const float*)d_in[8];
  const float* w_c2   = (const float*)d_in[9];
  const float* b_c2   = (const float*)d_in[10];
  float* out = (float*)d_out;
  float* ws  = (float*)d_ws;

  k_down_partial<<<256, 256, 0, stream>>>(x, w_down, ws + WS_PART);
  k_finalize<<<1, 256, 0, stream>>>(ws + WS_PART, b_down, w_hash, b_hash,
                                    w_c1, b_c1, w_c2, ws, out + OFF_BINS);
  k_up<<<NEL / 1024, 256, 0, stream>>>(w_up, b_up, ws, out + OFF_XUP);
  k_centroid<<<NEL / 128, 256, 0, stream>>>(x, ws, w_c1, b_c2, out);
}
